// FINN_Burger_71425306132782
// MI455X (gfx1250) — matrix operand order repaired and checked by reading, not compiled, not run on hardware
//
#include <hip/hip_runtime.h>

// FINN Burgers neural-ODE, RK4, fused one-RK4-step-per-kernel for gfx1250.
// GEMM [Nx x 128] @ [128 x 128] per stage via v_wmma_f32_16x16x32_f16.
// tanh via hardware v_tanh_f32 when the builtin is available.

typedef __attribute__((ext_vector_type(16))) _Float16 v16h;
typedef __attribute__((ext_vector_type(8)))  float    v8f;

#define NXC   65536
#define HC    128
#define TM    256                 // cells per block
#define HALO  16                  // >= 4 needed; 16 = one WMMA row-tile
#define NA    (TM + 2 * HALO)     // 288 rows get an MLP evaluation
#define NWAVES (NA / 16)          // 18 waves, one 16-row tile each
#define NT    (NWAVES * 32)       // 576 threads

__device__ __forceinline__ float fast_tanh(float x) {
#if __has_builtin(__builtin_amdgcn_tanhf)
  return __builtin_amdgcn_tanhf(x);               // v_tanh_f32 (1 trans op)
#else
  x = fminf(fmaxf(x, -10.0f), 10.0f);
  float e = __expf(2.0f * x);                     // v_exp_f32
  return (e - 1.0f) * __builtin_amdgcn_rcpf(e + 1.0f);
#endif
}

union AFrag { v16h v; _Float16 e[16]; };
union BFrag { v16h v; unsigned int u[8]; };
union Pack2 { _Float16 h[2]; unsigned int u; };

__global__ void copy_row0(const float* __restrict__ u0, float* __restrict__ out) {
  int i = blockIdx.x * blockDim.x + threadIdx.x;
  if (i < NXC) out[i] = u0[i];
}

__global__ __launch_bounds__(NT) void finn_rk4_step(
    const float* __restrict__ u_in,   // u at time t   (row of d_out)
    float* __restrict__ u_out,        // u at time t+1 (next row of d_out)
    const float* __restrict__ tg, int step,
    const float* __restrict__ W1, const float* __restrict__ W2,
    const float* __restrict__ W3, const float* __restrict__ Dp,
    const float* __restrict__ BC)
{
  // W2 pre-swizzled into WMMA B-fragment order: [kb][nt][lane*8 + vgpr]
  __shared__ unsigned int w2frag[4][8][32 * 8];   // 32 KB
  __shared__ float w1s[HC], w3s[HC];
  __shared__ float ubase[NA], us[NA], as_[NA];
  __shared__ float ks[4][NA];

  const int tid  = threadIdx.x;
  const int lane = tid & 31;
  const int wv   = tid >> 5;
  const int cb   = blockIdx.x * TM;

  for (int i = tid; i < HC; i += NT) { w1s[i] = W1[i]; w3s[i] = W3[i]; }

  // Swizzle W2 (row-major [k][n]) into f16 B fragments.
  // B 32x16 f16 layout: lanes 0-15 -> n = nt*16+lane, K=0..7,16..23 (2/vgpr);
  //                     lanes 16-31 -> same n, K=8..15,24..31.
  for (int idx = tid; idx < 4 * 8 * 32 * 8; idx += NT) {
    int v  = idx & 7;
    int l  = (idx >> 3) & 31;
    int nt = (idx >> 8) & 7;
    int kb = idx >> 11;
    int n  = nt * 16 + (l & 15);
    int k  = kb * 32 + ((v < 4) ? (((l < 16) ? 0 : 8) + 2 * v) : (16 + ((l < 16) ? 0 : 8) + 2 * (v - 4)));
    Pack2 p;
    p.h[0] = (_Float16)W2[k * HC + n];
    p.h[1] = (_Float16)W2[(k + 1) * HC + n];
    w2frag[kb][nt][l * 8 + v] = p.u;
  }

  for (int i = tid; i < NA; i += NT) {
    int g = cb - HALO + i;
    float v = (g >= 0 && g < NXC) ? u_in[g] : 0.0f;
    ubase[i] = v;
    us[i]    = v;
  }
  __syncthreads();

  const float dt  = tg[step + 1] - tg[step];
  const float d   = Dp[0];
  const float bc0 = BC[0];
  const float bc1 = BC[1];

  const int rb    = wv * 16;        // this wave's local row base
  const int lh    = lane & 15;
  const int hf    = lane >> 4;      // 0: rows rb..rb+7 (D), 1: rows rb+8..rb+15
  const int khalf = hf * 8;         // A-fragment K-half select

  for (int s = 0; s < 4; ++s) {
    // ---- per-cell MLP for 16 rows per wave: a = W3ᵀ·tanh(W2ᵀ·tanh(W1ᵀ·u)) ----
    v8f acc[8];
    #pragma unroll
    for (int nt = 0; nt < 8; ++nt)
      #pragma unroll
      for (int j = 0; j < 8; ++j) acc[nt][j] = 0.0f;

    float uv = us[rb + lh];         // row value for this lane (A layout: M = lane&15)

    #pragma unroll
    for (int kb = 0; kb < 4; ++kb) {
      // Layer 1 straight into the 16-bit A 16x32 fragment layout.
      AFrag af;
      #pragma unroll
      for (int v = 0; v < 8; ++v) {
        int k = kb * 32 + ((v < 4) ? (khalf + 2 * v) : (16 + khalf + 2 * (v - 4)));
        af.e[2 * v]     = (_Float16)fast_tanh(uv * w1s[k]);
        af.e[2 * v + 1] = (_Float16)fast_tanh(uv * w1s[k + 1]);
      }
      #pragma unroll
      for (int nt = 0; nt < 8; ++nt) {
        BFrag bf;
        #pragma unroll
        for (int v = 0; v < 8; ++v) bf.u[v] = w2frag[kb][nt][lane * 8 + v];
        acc[nt] = __builtin_amdgcn_wmma_f32_16x16x32_f16(
            false, af.v, false, bf.v, (short)0, acc[nt], false, false);
      }
    }

    // Layer 3 on the D-fragment layout (lane n = nt*16+lh, vgpr v -> row rb+hf*8+v),
    // then 16-lane xor-tree reduction over n.
    #pragma unroll
    for (int v = 0; v < 8; ++v) {
      float p = 0.0f;
      #pragma unroll
      for (int nt = 0; nt < 8; ++nt)
        p += fast_tanh(acc[nt][v]) * w3s[nt * 16 + lh];
      p += __shfl_xor(p, 1, 32);
      p += __shfl_xor(p, 2, 32);
      p += __shfl_xor(p, 4, 32);
      p += __shfl_xor(p, 8, 32);
      if (lh == 0) as_[rb + hf * 8 + v] = p;
    }
    __syncthreads();

    // ---- stencil flux: f[i] = (d+relu(a))*(uL-uc) + (d+relu(-a))*(uR-uc) ----
    for (int i = tid + 1; i < NA - 1; i += NT) {
      int g = cb - HALO + i;
      float uc = us[i];
      float uL = (g == 0)       ? bc0 : us[i - 1];
      float uR = (g == NXC - 1) ? bc1 : us[i + 1];
      float a  = as_[i];
      float ap = fmaxf(a, 0.0f);
      float am = fmaxf(-a, 0.0f);
      ks[s][i] = (d + ap) * (uL - uc) + (d + am) * (uR - uc);
    }
    __syncthreads();

    if (s < 3) {
      float c = (s < 2) ? 0.5f * dt : dt;
      for (int i = tid + 1; i < NA - 1; i += NT)
        us[i] = ubase[i] + c * ks[s][i];
      __syncthreads();
    }
  }

  // ---- RK4 combine for the block's own 256 cells ----
  for (int i = tid; i < TM; i += NT) {
    int ia = HALO + i;
    u_out[cb + i] = ubase[ia] + (dt * (1.0f / 6.0f)) *
        (ks[0][ia] + 2.0f * ks[1][ia] + 2.0f * ks[2][ia] + ks[3][ia]);
  }
}

extern "C" void kernel_launch(void* const* d_in, const int* in_sizes, int n_in,
                              void* d_out, int out_size, void* d_ws, size_t ws_size,
                              hipStream_t stream) {
  (void)n_in; (void)out_size; (void)d_ws; (void)ws_size;
  const float* tg = (const float*)d_in[0];
  const float* u0 = (const float*)d_in[1];
  const float* W1 = (const float*)d_in[2];
  const float* W2 = (const float*)d_in[3];
  const float* W3 = (const float*)d_in[4];
  const float* D  = (const float*)d_in[5];
  const float* BC = (const float*)d_in[6];
  float* out = (float*)d_out;

  const int T = in_sizes[0];

  copy_row0<<<(NXC + 255) / 256, 256, 0, stream>>>(u0, out);
  for (int s = 0; s < T - 1; ++s) {
    finn_rk4_step<<<NXC / TM, NT, 0, stream>>>(
        out + (size_t)s * NXC, out + (size_t)(s + 1) * NXC,
        tg, s, W1, W2, W3, D, BC);
  }
}
